// BiLSTMw2v_75127567941889
// MI455X (gfx1250) — matrix operand order repaired and checked by reading, not compiled, not run on hardware
//
#include <hip/hip_runtime.h>
#include <stdint.h>

#define TLEN  8192
#define EMB   300
#define EK    320          // EMB padded to multiple of 32 (bf16 WMMA K)
#define HID   200
#define G4    800          // 4*HID
#define NEXTRA 50
#define OUTD  2
#define MT    (TLEN/16)    // 512 row tiles
#define NT    (G4/16)      // 50 col tiles
#define KB    (EK/32)      // 10 K-steps

typedef __attribute__((ext_vector_type(16))) __bf16 v16bf;
typedef __attribute__((ext_vector_type(8)))  float  v8f;

union V16BF { v16bf v; uint4 q[2]; };

__device__ __forceinline__ unsigned short f2bf(float f) {
    unsigned int u = __float_as_uint(f);
    u += 0x7FFFu + ((u >> 16) & 1u);      // round-to-nearest-even
    return (unsigned short)(u >> 16);
}

__device__ __forceinline__ float sigm(float x)      { return 1.0f / (1.0f + __expf(-x)); }
__device__ __forceinline__ float tanh_fast(float x) { return 2.0f / (1.0f + __expf(-2.0f * x)) - 1.0f; }

// ---------------------------------------------------------------------------
// 1) Embedding gather + mask + relu -> bf16, K padded 300->320 with zeros
// ---------------------------------------------------------------------------
__global__ void embed_kernel(const int* __restrict__ x,
                             const float* __restrict__ emb,
                             unsigned short* __restrict__ ebf) {
    int idx = blockIdx.x * blockDim.x + threadIdx.x;   // [0, TLEN*EK)
    int t = idx / EK;
    int k = idx - t * EK;
    float v = 0.0f;
    int tok = x[t];
    if (k < EMB && tok != -1) {
        float e = emb[(size_t)tok * EMB + k];
        v = e > 0.0f ? e : 0.0f;                       // relu (masked rows stay 0)
    }
    ebf[idx] = f2bf(v);
}

// ---------------------------------------------------------------------------
// 2) Wih (both directions) -> bf16, K padded 300->320
// ---------------------------------------------------------------------------
__global__ void wconv_kernel(const float* __restrict__ Wih_f,
                             const float* __restrict__ Wih_b,
                             unsigned short* __restrict__ wbf) {
    int idx = blockIdx.x * blockDim.x + threadIdx.x;   // [0, 2*G4*EK)
    int dir = idx / (G4 * EK);
    int r   = idx - dir * (G4 * EK);
    int j   = r / EK;
    int k   = r - j * EK;
    const float* W = dir ? Wih_b : Wih_f;
    float v = (k < EMB) ? W[j * EMB + k] : 0.0f;
    wbf[idx] = f2bf(v);
}

// ---------------------------------------------------------------------------
// 3) ex[dir] = e @ Wih[dir].T + b[dir]   via v_wmma_f32_16x16x32_bf16
//    one wave = one 16x16 tile of ex; grid covers both directions exactly.
// ---------------------------------------------------------------------------
__global__ __launch_bounds__(256) void exgemm_kernel(
        const unsigned short* __restrict__ ebf,
        const unsigned short* __restrict__ wbf,
        const float* __restrict__ b_f,
        const float* __restrict__ b_b,
        float* __restrict__ ex) {
    int wave = blockIdx.x * (blockDim.x >> 5) + (threadIdx.x >> 5);
    int lane = threadIdx.x & 31;
    int dir  = wave / (MT * NT);
    int rem  = wave - dir * (MT * NT);
    int mt   = rem / NT;
    int nt   = rem - mt * NT;

    int half = lane >> 4;            // lane group (0: lanes 0-15, 1: lanes 16-31)
    int l15  = lane & 15;

    // A-fragment (16x32 bf16): lane holds row M=l15; K runs {half*8+0..7, half*8+16..23}
    const unsigned short* arow = ebf + (size_t)(mt * 16 + l15) * EK + half * 8;
    // B-fragment (32x16 bf16): lane holds col N=l15 (= Wih row j); K runs {half*8+0..7, half*8+16..23}
    const unsigned short* brow = wbf + (size_t)dir * (G4 * EK)
                                     + (size_t)(nt * 16 + l15) * EK + half * 8;

    v8f acc = {0.f, 0.f, 0.f, 0.f, 0.f, 0.f, 0.f, 0.f};
    #pragma unroll
    for (int kb = 0; kb < KB; ++kb) {
        V16BF a, b;
        a.q[0] = *(const uint4*)(arow + kb * 32);        // K = kb*32 + half*8 + 0..7
        a.q[1] = *(const uint4*)(arow + kb * 32 + 16);   // K = kb*32 + half*8 + 16..23
        b.q[0] = *(const uint4*)(brow + kb * 32);        // K = kb*32 + half*8 + 0..7
        b.q[1] = *(const uint4*)(brow + kb * 32 + 16);   // K = kb*32 + half*8 + 16..23
        acc = __builtin_amdgcn_wmma_f32_16x16x32_bf16(
                  false, a.v, false, b.v, (short)0, acc, false, false);
    }

    int j = nt * 16 + l15;                               // output column (gate index)
    const float* bvec = dir ? b_b : b_f;
    float bias = bvec[j];
    float* exd = ex + (size_t)dir * TLEN * G4;
    int mbase = mt * 16 + half * 8;                      // D: VGPR r -> M = mbase + r
    #pragma unroll
    for (int r = 0; r < 8; ++r)
        exd[(size_t)(mbase + r) * G4 + j] = acc[r] + bias;
}

// ---------------------------------------------------------------------------
// 4) Sequential LSTM scan: blockIdx.x = direction, 800 threads = one gate each
// ---------------------------------------------------------------------------
__global__ __launch_bounds__(800) void scan_kernel(
        const float* __restrict__ ex,
        const float* __restrict__ Whh_f,
        const float* __restrict__ Whh_b,
        float* __restrict__ hfor,
        float* __restrict__ hback) {
    __shared__ float hsh[HID];
    __shared__ float gsh[G4];
    int dir = blockIdx.x;
    int tid = threadIdx.x;
    const float* Whh = dir ? Whh_b : Whh_f;
    const float* exd = ex + (size_t)dir * TLEN * G4;
    float* outp = dir ? hback : hfor;

    if (tid < HID) hsh[tid] = 0.0f;
    float c = 0.0f;
    __syncthreads();

    const float* wrow = Whh + tid * HID;                 // row j of Whh, contiguous
    for (int s = 0; s < TLEN; ++s) {
        int t = dir ? (TLEN - 1 - s) : s;
        float acc = exd[(size_t)t * G4 + tid];
        if (s + 1 < TLEN) {
            int tn = dir ? (t - 1) : (t + 1);
            __builtin_prefetch(exd + (size_t)tn * G4 + tid, 0, 3);
        }
        #pragma unroll 5
        for (int k = 0; k < HID; k += 4) {
            float4 w = *(const float4*)(wrow + k);
            acc += w.x * hsh[k]     + w.y * hsh[k + 1]
                 + w.z * hsh[k + 2] + w.w * hsh[k + 3];
        }
        gsh[tid] = acc;
        __syncthreads();
        if (tid < HID) {
            float ig = sigm(gsh[tid]);                   // PyTorch gate order i,f,g,o
            float fg = sigm(gsh[HID + tid]);
            float gg = tanh_fast(gsh[2 * HID + tid]);
            float og = sigm(gsh[3 * HID + tid]);
            c = fg * c + ig * gg;
            float h = og * tanh_fast(c);
            hsh[tid] = h;
            outp[(size_t)t * HID + tid] = h;
        }
        __syncthreads();
    }
}

// ---------------------------------------------------------------------------
// 5) Fused head: s = relu([h_for,h_back] @ W_h2s.T + b), out = s @ W_s2o.T + b
// ---------------------------------------------------------------------------
__global__ __launch_bounds__(256) void head_kernel(
        const float* __restrict__ hfor,
        const float* __restrict__ hback,
        const float* __restrict__ W_h2s,
        const float* __restrict__ b_h2s,
        const float* __restrict__ W_s2o,
        const float* __restrict__ b_s2o,
        float* __restrict__ outp) {
    int t = blockIdx.x * blockDim.x + threadIdx.x;
    float acc[NEXTRA];
    #pragma unroll
    for (int u = 0; u < NEXTRA; ++u) acc[u] = b_h2s[u];

    const float* hf = hfor  + (size_t)t * HID;
    const float* hb = hback + (size_t)t * HID;
    for (int k = 0; k < HID; ++k) {
        float hk = hf[k];
        #pragma unroll
        for (int u = 0; u < NEXTRA; ++u) acc[u] += hk * W_h2s[u * 2 * HID + k];
    }
    for (int k = 0; k < HID; ++k) {
        float hk = hb[k];
        #pragma unroll
        for (int u = 0; u < NEXTRA; ++u) acc[u] += hk * W_h2s[u * 2 * HID + HID + k];
    }
    #pragma unroll
    for (int v = 0; v < OUTD; ++v) {
        float o = b_s2o[v];
        #pragma unroll
        for (int u = 0; u < NEXTRA; ++u) {
            float su = acc[u] > 0.0f ? acc[u] : 0.0f;
            o += W_s2o[v * NEXTRA + u] * su;
        }
        outp[(size_t)t * OUTD + v] = o;
    }
}

extern "C" void kernel_launch(void* const* d_in, const int* in_sizes, int n_in,
                              void* d_out, int out_size, void* d_ws, size_t ws_size,
                              hipStream_t stream) {
    (void)in_sizes; (void)n_in; (void)out_size; (void)ws_size;
    const int*   x     = (const int*)  d_in[0];
    const float* emb   = (const float*)d_in[1];
    const float* Wih_f = (const float*)d_in[2];
    const float* Whh_f = (const float*)d_in[3];
    const float* b_f   = (const float*)d_in[4];
    const float* Wih_b = (const float*)d_in[5];
    const float* Whh_b = (const float*)d_in[6];
    const float* b_b   = (const float*)d_in[7];
    const float* W_h2s = (const float*)d_in[8];
    const float* b_h2s = (const float*)d_in[9];
    const float* W_s2o = (const float*)d_in[10];
    const float* b_s2o = (const float*)d_in[11];
    float* outp = (float*)d_out;

    // workspace carve-up (~72 MB total), 256B aligned chunks
    char* ws = (char*)d_ws;
    size_t off = 0;
    auto carve = [&](size_t bytes) {
        void* p = ws + off;
        off = (off + bytes + 255) & ~(size_t)255;
        return p;
    };
    unsigned short* ebf  = (unsigned short*)carve((size_t)TLEN * EK * 2);      //  5.2 MB
    unsigned short* wbf  = (unsigned short*)carve((size_t)2 * G4 * EK * 2);    //  1.0 MB
    float*          ex   = (float*)carve((size_t)2 * TLEN * G4 * 4);           // 52.4 MB
    float*          hfor = (float*)carve((size_t)TLEN * HID * 4);              //  6.6 MB
    float*          hbak = (float*)carve((size_t)TLEN * HID * 4);              //  6.6 MB

    embed_kernel <<<(TLEN * EK) / 256,   256, 0, stream>>>(x, emb, ebf);
    wconv_kernel <<<(2 * G4 * EK) / 256, 256, 0, stream>>>(Wih_f, Wih_b, wbf);
    exgemm_kernel<<<(2 * MT * NT) / 8,   256, 0, stream>>>(ebf, wbf, b_f, b_b, ex);
    scan_kernel  <<<2,                   800, 0, stream>>>(ex, Whh_f, Whh_b, hfor, hbak);
    head_kernel  <<<TLEN / 256,          256, 0, stream>>>(hfor, hbak, W_h2s, b_h2s,
                                                           W_s2o, b_s2o, outp);
}
